// GATModule_16810501997067
// MI455X (gfx1250) — compile-verified, matrix operand order checked by reading, not run on hardware
//
#include <hip/hip_runtime.h>
#include <hip/hip_bf16.h>
#include <math.h>

typedef _Float16 v16h __attribute__((ext_vector_type(16)));
typedef _Float16 v8h  __attribute__((ext_vector_type(8)));
typedef _Float16 v4h  __attribute__((ext_vector_type(4)));
typedef float    v8f  __attribute__((ext_vector_type(8)));

#define HH 128
#define BB 4096
#define KK 32

// ---------------------------------------------------------------------------
// half-wave (16-lane group) sum: masks 1,2,4,8 stay within lanes [0,16) / [16,32)
__device__ __forceinline__ float halfwave_sum(float s) {
  s += __shfl_xor(s, 1, 32);
  s += __shfl_xor(s, 2, 32);
  s += __shfl_xor(s, 4, 32);
  s += __shfl_xor(s, 8, 32);
  return s;
}

// convert 4 consecutive f32 to packed f16 and store 8B to LDS
__device__ __forceinline__ void cvt_store4(_Float16* dst, float4 f) {
  v4h h = {(_Float16)f.x, (_Float16)f.y, (_Float16)f.z, (_Float16)f.w};
  *(v4h*)dst = h;
}

// ---------------------------------------------------------------------------
// Wave-level GEMM: 16 rows (f16, staged in LDS, row-major, leading dim ldk)
// times a pre-swizzled f16 weight matrix (B-fragments contiguous per lane),
// producing 16x128 fp32 in acc[8] (each v8f = one 16x16 D tile).
// A-fragment layout (16-bit A 16x32, ISA 7.12.2): M = lane%16,
//   lanes 0-15:  halves j=0..7 -> K=0..7,  j=8..15 -> K=16..23
//   lanes 16-31: halves j=0..7 -> K=8..15, j=8..15 -> K=24..31
template<int KTILES>
__device__ __forceinline__ void wave_gemm(const _Float16* xs, int ldk,
                                          const _Float16* pw, int lane,
                                          v8f acc[8]) {
#pragma unroll
  for (int kt = 0; kt < KTILES; ++kt) {
    const int m  = lane & 15;
    const int k0 = kt * 32 + ((lane & 16) ? 8 : 0);
    v8h lo = *(const v8h*)(xs + m * ldk + k0);
    v8h hi = *(const v8h*)(xs + m * ldk + k0 + 16);
    v16h a = __builtin_shufflevector(lo, hi, 0,1,2,3,4,5,6,7,8,9,10,11,12,13,14,15);
#pragma unroll
    for (int nt = 0; nt < 8; ++nt) {
      v16h b = *(const v16h*)(pw + ((size_t)(kt * 8 + nt) * 32 + lane) * 16);
      acc[nt] = __builtin_amdgcn_wmma_f32_16x16x32_f16(
          false, a, false, b, (short)0, acc[nt], false, false);
    }
  }
}

__device__ __forceinline__ void zero_acc(v8f acc[8]) {
  v8f z = {0.f,0.f,0.f,0.f,0.f,0.f,0.f,0.f};
#pragma unroll
  for (int i = 0; i < 8; ++i) acc[i] = z;
}

// bias + ReLU + LayerNorm over the 16x128 tile held in acc[8].
// D layout: VGPR r holds (M=r, N=lane) for lanes 0-15 and (M=r+8, N=lane-16)
// for lanes 16-31 -> rows reduce across the 16-lane half via halfwave_sum.
__device__ __forceinline__ void bias_relu_ln(v8f acc[8], int lane,
                                             const float* bias, const float* g,
                                             const float* be) {
  float bi[8], gg[8], bb[8];
#pragma unroll
  for (int nt = 0; nt < 8; ++nt) {
    int n = nt * 16 + (lane & 15);
    bi[nt] = bias[n]; gg[nt] = g[n]; bb[nt] = be[n];
  }
#pragma unroll
  for (int nt = 0; nt < 8; ++nt)
#pragma unroll
    for (int r = 0; r < 8; ++r)
      acc[nt][r] = fmaxf(acc[nt][r] + bi[nt], 0.f);

  float mean[8], inv[8];
#pragma unroll
  for (int r = 0; r < 8; ++r) {
    float s = 0.f;
#pragma unroll
    for (int nt = 0; nt < 8; ++nt) s += acc[nt][r];
    mean[r] = halfwave_sum(s) * (1.f / 128.f);
  }
#pragma unroll
  for (int r = 0; r < 8; ++r) {
    float s = 0.f;
#pragma unroll
    for (int nt = 0; nt < 8; ++nt) { float d = acc[nt][r] - mean[r]; s += d * d; }
    inv[r] = rsqrtf(halfwave_sum(s) * (1.f / 128.f) + 1e-5f);
  }
#pragma unroll
  for (int nt = 0; nt < 8; ++nt)
#pragma unroll
    for (int r = 0; r < 8; ++r)
      acc[nt][r] = gg[nt] * (acc[nt][r] - mean[r]) * inv[r] + bb[nt];
}

// ---------------------------------------------------------------------------
// Convert fp32 weights to f16 pre-swizzled into B-fragment order so a lane's
// fragment is one contiguous 32B load: dst[(f*32+lane)*16 + j], f = kt*8+nt.
__global__ void k_prepack(const float* Wa, const float* W1, const float* W2,
                          _Float16* pWaTop, _Float16* pWaBot,
                          _Float16* pW1, _Float16* pW2) {
  int t = blockIdx.x * blockDim.x + threadIdx.x;
  const float* src; int rowoff; _Float16* dst; int local;
  if      (t < 1024) { src = Wa; rowoff = 0;   dst = pWaTop; local = t;        }
  else if (t < 2048) { src = Wa; rowoff = 128; dst = pWaBot; local = t - 1024; }
  else if (t < 4096) { src = W1; rowoff = 0;   dst = pW1;    local = t - 2048; }
  else if (t < 5120) { src = W2; rowoff = 0;   dst = pW2;    local = t - 4096; }
  else return;
  int lane = local & 31, f = local >> 5, nt = f & 7, kt = f >> 3;
  int n = nt * 16 + (lane & 15);
  v16h v;
#pragma unroll
  for (int j = 0; j < 16; ++j) {
    int k = kt * 32 + ((j < 8) ? j : j + 8) + ((lane & 16) ? 8 : 0);
    v[j] = (_Float16)src[(size_t)(rowoff + k) * HH + n];
  }
  *(v16h*)(dst + (size_t)local * 16) = v;
}

// ---------------------------------------------------------------------------
// qproj[R] = this_emb[R] @ Wa_top + ba  (rows 0..B-1: users, B..2B-1: items)
__global__ void k_qproj(const float* user_emb, const float* item_emb,
                        const int* users_ind, const int* items_ind,
                        const float* ba, const _Float16* pWaTop, float* qproj) {
  __shared__ __align__(16) _Float16 xs[4 * 16 * HH];
  const int lane = threadIdx.x & 31, wid = threadIdx.x >> 5;
  const int r0 = (blockIdx.x * 4 + wid) * 16;
  _Float16* stage = xs + wid * 16 * HH;
  {
    int m = lane >> 1, R = r0 + m;
    const float* src = (R < BB) ? (user_emb + (size_t)users_ind[R] * HH)
                                : (item_emb + (size_t)items_ind[R - BB] * HH);
    int c0 = (lane & 1) * 64;
    const float4* s4 = (const float4*)(src + c0);
#pragma unroll
    for (int j = 0; j < 16; ++j)
      cvt_store4(stage + m * HH + c0 + 4 * j, s4[j]);
  }
  __syncthreads();
  v8f acc[8]; zero_acc(acc);
  wave_gemm<4>(stage, HH, pWaTop, lane, acc);
  const int half8 = (lane & 16) ? 8 : 0;
#pragma unroll
  for (int nt = 0; nt < 8; ++nt) {
    int n = nt * 16 + (lane & 15);
    float b = ba[n];
#pragma unroll
    for (int r = 0; r < 8; ++r)
      qproj[(size_t)(r0 + r + half8) * HH + n] = acc[nt][r] + b;
  }
}

// ---------------------------------------------------------------------------
// attention logits: each wave owns (b, quarter q) -> 16 neighbor rows.
// key rows: n<32 -> review_emb[relInd], n>=32 -> sameEmb[sameInd] * this.
// e[b,n] = va . tanh(key @ Wa_bot + qproj[b])
__global__ void k_attn(const float* thisEmb, const int* thisInd,
                       const float* relEmb, const int* relInd,
                       const float* sameEmb, const int* sameInd,
                       const float* qproj_side, const float* va,
                       const _Float16* pWaBot, float* e_out) {
  __shared__ __align__(16) _Float16 xs[4 * 16 * HH];
  const int lane = threadIdx.x & 31, wid = threadIdx.x >> 5;
  const int w = blockIdx.x * 4 + wid;
  const int b = w >> 2, q = w & 3;
  _Float16* stage = xs + wid * 16 * HH;
  const float* thisRow = thisEmb + (size_t)thisInd[b] * HH;
  {
    int m = lane >> 1, n = q * 16 + m;
    int c0 = (lane & 1) * 64;
    if (n < 32) {
      const float4* s4 = (const float4*)(relEmb + (size_t)relInd[b * KK + n] * HH + c0);
#pragma unroll
      for (int j = 0; j < 16; ++j)
        cvt_store4(stage + m * HH + c0 + 4 * j, s4[j]);
    } else {
      const float4* s4 = (const float4*)(sameEmb + (size_t)sameInd[b * KK + (n - 32)] * HH + c0);
      const float4* t4 = (const float4*)(thisRow + c0);
#pragma unroll
      for (int j = 0; j < 16; ++j) {
        float4 a = s4[j], t = t4[j];
        float4 p = make_float4(a.x * t.x, a.y * t.y, a.z * t.z, a.w * t.w);
        cvt_store4(stage + m * HH + c0 + 4 * j, p);
      }
    }
  }
  __syncthreads();
  v8f acc[8]; zero_acc(acc);
  wave_gemm<4>(stage, HH, pWaBot, lane, acc);
  float qp[8], vv[8];
#pragma unroll
  for (int nt = 0; nt < 8; ++nt) {
    int n = nt * 16 + (lane & 15);
    qp[nt] = qproj_side[(size_t)b * HH + n];
    vv[nt] = va[n];
  }
  const int half8 = (lane & 16) ? 8 : 0;
#pragma unroll
  for (int r = 0; r < 8; ++r) {
    float p = 0.f;
#pragma unroll
    for (int nt = 0; nt < 8; ++nt)
      p += tanhf(acc[nt][r] + qp[nt]) * vv[nt];
    p = halfwave_sum(p);
    if ((lane & 15) == 0)
      e_out[(size_t)b * 64 + q * 16 + r + half8] = p;
  }
}

// ---------------------------------------------------------------------------
// softmax over 64 logits + weighted sum of value embeddings.
// Value rows are 512B = 32 lanes x 16B -> one global_load_async_to_lds_b128
// per row per wave (ASYNCcnt-tracked, no VGPR returns); softmax weights are
// computed while the copies are in flight, then s_wait_asynccnt + barrier.
__global__ void k_soft(const float* e_in, const int* diffInd, const int* sameInd,
                       const float* diffEmb, const float* sameEmb,
                       float* pref_out) {
  __shared__ __align__(16) float sVal[64 * HH];  // 32 KB gathered value rows
  __shared__ float sW[64];
  __shared__ int   sI[64];
  __shared__ float sM, sS;
  const int b = blockIdx.x, t = threadIdx.x;  // blockDim = 128 = H
  const int lane = t & 31, wid = t >> 5;
  if (t < 64) {
    sW[t] = e_in[(size_t)b * 64 + t];
    sI[t] = (t < 32) ? diffInd[b * KK + t] : sameInd[b * KK + t - 32];
  }
  __syncthreads();
  // each wave async-copies 16 value rows into LDS
#pragma unroll 4
  for (int n = wid * 16; n < wid * 16 + 16; ++n) {
    const float* rowp = ((n < 32) ? diffEmb : sameEmb) + (size_t)sI[n] * HH;
    const char* ga = (const char*)rowp + lane * 16;
    // flat address of LDS: low 32 bits == LDS byte offset (aperture rule)
    unsigned la = (unsigned)(size_t)(&sVal[n * HH]) + lane * 16;
    asm volatile("global_load_async_to_lds_b128 %0, %1, off"
                 :: "v"(la), "v"(ga) : "memory");
  }
  // softmax over the 64 logits while the async copies are in flight
  if (t == 0) {
    float m = sW[0];
    for (int n = 1; n < 64; ++n) m = fmaxf(m, sW[n]);
    sM = m;
  }
  __syncthreads();
  if (t < 64) sW[t] = expf(sW[t] - sM);
  __syncthreads();
  if (t == 0) {
    float s = 0.f;
    for (int n = 0; n < 64; ++n) s += sW[n];
    sS = 1.f / s;
  }
  asm volatile("s_wait_asynccnt 0x0" ::: "memory");
  __syncthreads();
  float acc = 0.f;
#pragma unroll 8
  for (int n = 0; n < 64; ++n)
    acc += sW[n] * sVal[n * HH + t];
  pref_out[(size_t)b * HH + t] = acc * sS;
}

// ---------------------------------------------------------------------------
// transform: x=[this,pref] (2H) -> ReLU(x@W1+b1) -> LN -> ReLU(@W2+b2) -> LN
__global__ void k_transform(const float* user_emb, const float* item_emb,
                            const int* users_ind, const int* items_ind,
                            const float* pref,
                            const _Float16* pW1, const _Float16* pW2,
                            const float* b1, const float* g1, const float* be1,
                            const float* b2, const float* g2, const float* be2,
                            float* out) {
  __shared__ __align__(16) _Float16 xs[4 * 16 * 256];
  __shared__ __align__(16) _Float16 ys[4 * 16 * 128];
  const int lane = threadIdx.x & 31, wid = threadIdx.x >> 5;
  const int r0 = (blockIdx.x * 4 + wid) * 16;
  _Float16* sx = xs + wid * 16 * 256;
  _Float16* sy = ys + wid * 16 * 128;
  {
    int m = lane >> 1, half = lane & 1, R = r0 + m;
    const float* src;
    if (half == 0) {
      src = (R < BB) ? user_emb + (size_t)users_ind[R] * HH
                     : item_emb + (size_t)items_ind[R - BB] * HH;
    } else {
      src = pref + (size_t)R * HH;
    }
    const float4* s4 = (const float4*)src;
    _Float16* dst = sx + m * 256 + half * HH;
#pragma unroll
    for (int j = 0; j < 32; ++j)
      cvt_store4(dst + 4 * j, s4[j]);
  }
  __syncthreads();
  v8f acc[8]; zero_acc(acc);
  wave_gemm<8>(sx, 256, pW1, lane, acc);
  bias_relu_ln(acc, lane, b1, g1, be1);

  const int half8 = (lane & 16) ? 8 : 0;
#pragma unroll
  for (int nt = 0; nt < 8; ++nt) {
    int n = nt * 16 + (lane & 15);
#pragma unroll
    for (int r = 0; r < 8; ++r)
      sy[(r + half8) * 128 + n] = (_Float16)acc[nt][r];
  }
  __syncthreads();
  zero_acc(acc);
  wave_gemm<4>(sy, 128, pW2, lane, acc);
  bias_relu_ln(acc, lane, b2, g2, be2);
#pragma unroll
  for (int nt = 0; nt < 8; ++nt) {
    int n = nt * 16 + (lane & 15);
#pragma unroll
    for (int r = 0; r < 8; ++r)
      out[(size_t)(r0 + r + half8) * HH + n] = acc[nt][r];
  }
}

// ---------------------------------------------------------------------------
__global__ void k_rel(float* out) {
  size_t i = (size_t)blockIdx.x * blockDim.x + threadIdx.x;
  const size_t N = (size_t)BB * HH;
  out[2 * N + i] = out[i] * out[N + i];
}

// ---------------------------------------------------------------------------
extern "C" void kernel_launch(void* const* d_in, const int* in_sizes, int n_in,
                              void* d_out, int out_size, void* d_ws, size_t ws_size,
                              hipStream_t stream) {
  (void)in_sizes; (void)n_in; (void)out_size; (void)ws_size;
  const int B = BB, H = HH;
  const int*   users_ind        = (const int*)d_in[0];
  const int*   items_ind        = (const int*)d_in[1];
  const int*   user_ne_items    = (const int*)d_in[2];
  const int*   user_ne_users    = (const int*)d_in[3];
  const int*   item_ne_users    = (const int*)d_in[4];
  const int*   item_ne_items    = (const int*)d_in[5];
  const int*   user_review_inds = (const int*)d_in[6];
  const int*   item_review_inds = (const int*)d_in[7];
  const float* user_emb   = (const float*)d_in[8];
  const float* item_emb   = (const float*)d_in[9];
  const float* review_emb = (const float*)d_in[10];
  const float* Wa  = (const float*)d_in[11];
  const float* ba  = (const float*)d_in[12];
  const float* va  = (const float*)d_in[13];
  const float* W1  = (const float*)d_in[14];
  const float* b1  = (const float*)d_in[15];
  const float* W2  = (const float*)d_in[16];
  const float* b2  = (const float*)d_in[17];
  const float* g1  = (const float*)d_in[18];
  const float* be1 = (const float*)d_in[19];
  const float* g2  = (const float*)d_in[20];
  const float* be2 = (const float*)d_in[21];
  float* out = (float*)d_out;

  char* w = (char*)d_ws;
  float* qproj = (float*)w; w += (size_t)2 * B * H * sizeof(float);
  float* e_ws  = (float*)w; w += (size_t)2 * B * 64 * sizeof(float);
  float* prefb = (float*)w; w += (size_t)2 * B * H * sizeof(float);
  _Float16* pWaTop = (_Float16*)w; w += 32 * 1024;
  _Float16* pWaBot = (_Float16*)w; w += 32 * 1024;
  _Float16* pW1    = (_Float16*)w; w += 64 * 1024;
  _Float16* pW2    = (_Float16*)w; w += 32 * 1024;

  k_prepack<<<20, 256, 0, stream>>>(Wa, W1, W2, pWaTop, pWaBot, pW1, pW2);
  k_qproj<<<128, 128, 0, stream>>>(user_emb, item_emb, users_ind, items_ind,
                                   ba, pWaTop, qproj);
  // user side
  k_attn<<<4096, 128, 0, stream>>>(user_emb, users_ind,
                                   review_emb, user_review_inds,
                                   user_emb, user_ne_users,
                                   qproj, va, pWaBot, e_ws);
  // item side
  k_attn<<<4096, 128, 0, stream>>>(item_emb, items_ind,
                                   review_emb, item_review_inds,
                                   item_emb, item_ne_items,
                                   qproj + (size_t)B * H, va, pWaBot,
                                   e_ws + (size_t)B * 64);
  k_soft<<<4096, 128, 0, stream>>>(e_ws, user_ne_items, user_ne_users,
                                   item_emb, user_emb, prefb);
  k_soft<<<4096, 128, 0, stream>>>(e_ws + (size_t)B * 64, item_ne_users,
                                   item_ne_items, user_emb, item_emb,
                                   prefb + (size_t)B * H);
  k_transform<<<128, 128, 0, stream>>>(user_emb, item_emb, users_ind, items_ind,
                                       prefb, pW1, pW2,
                                       b1, g1, be1, b2, g2, be2, out);
  k_rel<<<2048, 256, 0, stream>>>(out);
}
